// FasterRCNNWithFPN_19756849561694
// MI455X (gfx1250) — matrix operand order repaired and checked by reading, not compiled, not run on hardware
//
#include <hip/hip_runtime.h>

// ---------------------------------------------------------------------------
// Faster R-CNN w/ FPN forward pass for gfx1250 (MI455X), wave32 + WMMA f16.
// - All convs with Cin % 32 == 0 and all FC GEMMs use v_wmma_f32_16x16x32_f16.
// - fc1/fc2 use a 16x64-per-wave GEMM with the weight tile staged into LDS by
//   GLOBAL_LOAD_ASYNC_TO_LDS_B128 (double buffered, s_wait_asynccnt pipelined).
// ---------------------------------------------------------------------------

#define NB      2
#define IMGF    512.0f
#define NANCH   48384
#define SORTN   65536
#define PRENMS  2000
#define NPOS    660

#define OUT_RPNB (NB*NANCH)                 // 96768
#define OUT_FGO  (NB*NANCH*5)               // 483840
#define OUT_FGB  (OUT_FGO + NB*NPOS)        // 485160
#define OUT_CLS  (OUT_FGB + NB*NPOS*4)      // 490440
#define OUT_BOX  (OUT_CLS + NB*NPOS)        // 491760

typedef __attribute__((ext_vector_type(16))) _Float16 v16h;
typedef __attribute__((ext_vector_type(8)))  _Float16 v8h;
typedef __attribute__((ext_vector_type(8)))  float    v8f;

__device__ __forceinline__ int   iclamp(int v, int lo, int hi)   { return v < lo ? lo : (v > hi ? hi : v); }
__device__ __forceinline__ float fclampf(float v, float lo, float hi) { return fminf(fmaxf(v, lo), hi); }

// ---------------------------------------------------------------------------
// conv1: 3->64, 3x3 stride 2, SAME (pad_lo=0), relu.  K=27 not WMMA-friendly.
// ---------------------------------------------------------------------------
__global__ void conv1_kernel(const float* __restrict__ x,
                             const float* __restrict__ w,
                             float* __restrict__ y)
{
    int tid = blockIdx.x * blockDim.x + threadIdx.x;
    if (tid >= NB * 64 * 256 * 256) return;
    int px   = tid & 65535;
    int rest = tid >> 16;
    int oc   = rest & 63;
    int b    = rest >> 6;
    int oy = px >> 8, ox = px & 255;
    float acc = 0.0f;
    for (int ic = 0; ic < 3; ++ic)
        for (int ky = 0; ky < 3; ++ky) {
            int iy = oy * 2 + ky;
            if (iy >= 512) continue;
            for (int kx = 0; kx < 3; ++kx) {
                int ix = ox * 2 + kx;
                if (ix >= 512) continue;
                acc += x[(((size_t)b * 3 + ic) * 512 + iy) * 512 + ix] *
                       w[((oc * 3 + ic) * 3 + ky) * 3 + kx];
            }
        }
    y[((size_t)(b * 64 + oc) << 16) + px] = fmaxf(acc, 0.0f);
}

// ---------------------------------------------------------------------------
// Generic implicit-GEMM conv via WMMA f16.  One wave -> 16 oc x 16 pixels.
// A = weights (M=oc, K=ic within chunk), B = input (K=ic, N=pixel).
// Fragment layouts per CDNA5 ISA 7.12.2 (wave32).
// ---------------------------------------------------------------------------
__global__ __launch_bounds__(32) void conv_wmma_kernel(
    const float* __restrict__ x, const float* __restrict__ w,
    float* __restrict__ y,
    int Cin, int Cout, int Hin, int Win, int Hout, int Wout,
    int stride, int ksize, int pad, int relu)
{
    const int lane = threadIdx.x & 31;
    const int hh   = lane >> 4;     // lane half
    const int mn   = lane & 15;     // row (A) / col (B)
    const int p0   = blockIdx.x * 16;
    const int oc0  = blockIdx.y * 16;
    const int b    = blockIdx.z;
    const int HW     = Hout * Wout;
    const int HinWin = Hin * Win;
    const int ksz2   = ksize * ksize;

    const int p  = p0 + mn;
    const int oy = p / Wout;
    const int ox = p % Wout;
    const int ocA = oc0 + mn;
    const bool ain = (ocA < Cout);

    v8f acc = {};
    for (int tap = 0; tap < ksz2; ++tap) {
        const int ky = tap / ksize, kx = tap % ksize;
        const int iy = oy * stride + ky - pad;
        const int ix = ox * stride + kx - pad;
        const bool pin = (p < HW) && (iy >= 0) && (iy < Hin) && (ix >= 0) && (ix < Win);
        const float* xcol = x + ((size_t)b * Cin) * HinWin + (size_t)iy * Win + ix;
        const float* wrow = w + ((size_t)ocA * Cin) * ksz2 + tap;
        for (int k0 = 0; k0 < Cin; k0 += 32) {
            v16h a, bb;
#pragma unroll
            for (int e = 0; e < 16; ++e) {
                int kk = (e < 8) ? (hh * 8 + e) : (16 + hh * 8 + (e - 8));
                float wv = ain ? wrow[(size_t)(k0 + kk) * ksz2] : 0.0f;
                a[e] = (_Float16)wv;
                float xv = pin ? xcol[(size_t)(k0 + kk) * HinWin] : 0.0f;
                bb[e] = (_Float16)xv;
            }
            acc = __builtin_amdgcn_wmma_f32_16x16x32_f16(false, a, false, bb,
                                                         (short)0, acc, false, false);
        }
    }
#pragma unroll
    for (int r = 0; r < 8; ++r) {
        int oc = oc0 + hh * 8 + r;
        if (oc < Cout && p < HW) {
            float v = acc[r];
            if (relu) v = fmaxf(v, 0.0f);
            y[((size_t)b * Cout + oc) * HW + p] = v;
        }
    }
}

// ---------------------------------------------------------------------------
// Small GEMM: C[M,N] = A[M,K] f16 * Bw[N,K]^T f16, one 16x16 tile per wave.
// Used for the tiny cls (N=3) / box (N=12) heads.
// ---------------------------------------------------------------------------
__global__ __launch_bounds__(32) void gemm_wmma_kernel(
    const _Float16* __restrict__ A, const _Float16* __restrict__ Bw,
    float* __restrict__ Cf32, _Float16* __restrict__ Cf16,
    int M, int N, int K, int relu)
{
    const int lane = threadIdx.x & 31;
    const int hh   = lane >> 4;
    const int mn   = lane & 15;
    const int m0   = blockIdx.x * 16;
    const int n0   = blockIdx.y * 16;
    const bool ain = (m0 + mn) < M;
    const bool bin = (n0 + mn) < N;
    const _Float16* arow = A  + (size_t)(m0 + mn) * K;
    const _Float16* brow = Bw + (size_t)(n0 + mn) * K;

    v8f acc = {};
    for (int k0 = 0; k0 < K; k0 += 32) {
        v16h a, bb;
        if (ain) {
            a = *(const v16h*)(arow + k0 + hh * 16);
        } else {
#pragma unroll
            for (int e = 0; e < 16; ++e) a[e] = (_Float16)0.0f;
        }
        if (bin) {
            bb = *(const v16h*)(brow + k0 + hh * 16);
            __builtin_prefetch(brow + k0 + 128, 0, 1);
        } else {
#pragma unroll
            for (int e = 0; e < 16; ++e) bb[e] = (_Float16)0.0f;
        }
        acc = __builtin_amdgcn_wmma_f32_16x16x32_f16(false, a, false, bb,
                                                     (short)0, acc, false, false);
    }
#pragma unroll
    for (int r = 0; r < 8; ++r) {
        int mm = m0 + hh * 8 + r;
        int nn = n0 + mn;
        if (mm < M && nn < N) {
            float v = acc[r];
            if (relu) v = fmaxf(v, 0.0f);
            if (Cf32) Cf32[(size_t)mm * N + nn] = v;
            if (Cf16) Cf16[(size_t)mm * N + nn] = (_Float16)v;
        }
    }
}

// ---------------------------------------------------------------------------
// Big GEMM (N % 64 == 0, K % 32 == 0): one wave computes a 16x64 block.
// The 64-row x 32-half weight tile (4 KB) is staged into LDS with
// GLOBAL_LOAD_ASYNC_TO_LDS_B128 (ASYNCcnt), double buffered.  The A fragment
// is loaded once per K-chunk and reused by 4 WMMA ops.
// NOTE: lbuf is the only __shared__ object in this kernel, so its static LDS
// offset is 0; async-LDS asm operand uses the raw byte offset.
// ---------------------------------------------------------------------------
__global__ __launch_bounds__(32) void gemm64_wmma_kernel(
    const _Float16* __restrict__ A, const _Float16* __restrict__ Bw,
    _Float16* __restrict__ Cf16, int M, int N, int K, int relu)
{
    __shared__ alignas(16) _Float16 lbuf[2][64 * 32];   // 2 x 4 KB, LDS offset 0

    const int lane = threadIdx.x & 31;
    const int hh   = lane >> 4;
    const int mn   = lane & 15;
    const int m0   = blockIdx.x * 16;
    const int n0   = blockIdx.y * 64;
    const bool ain = (m0 + mn) < M;
    const _Float16* arow = A + (size_t)(m0 + mn) * K;
    const int nchunks = K >> 5;

    // stage 64x32-half tile (rows n0..n0+63, cols k0..k0+31) into lbuf[sel]
    auto stage = [&](int sel, int k0) {
#pragma unroll
        for (int q = 0; q < 8; ++q) {
            int ti      = q * 32 + lane;        // 0..255 transfers of 16B
            int row     = ti >> 2;              // 0..63
            int quarter = ti & 3;               // 0..3 (8 halves each)
            const _Float16* gsrc = Bw + (size_t)(n0 + row) * K + k0 + quarter * 8;
            unsigned ldsb = (unsigned)(sel * (64 * 32 * 2) + ti * 16);
            asm volatile("global_load_async_to_lds_b128 %0, %1, off"
                         :: "v"(ldsb), "v"(gsrc) : "memory");
        }
    };

    stage(0, 0);
    v8f z = {};
    v8f acc[4] = {z, z, z, z};

    for (int c = 0; c < nchunks; ++c) {
        const int k0 = c << 5;
        if (c + 1 < nchunks) {
            // all LDS reads of the buffer being overwritten have completed
            asm volatile("s_wait_dscnt 0x0" ::: "memory");
            stage((c + 1) & 1, k0 + 32);
            // previous chunk's 8 async transfers are done (8 newer in flight)
            asm volatile("s_wait_asynccnt 0x8" ::: "memory");
        } else {
            asm volatile("s_wait_asynccnt 0x0" ::: "memory");
        }

        v16h a;
        if (ain) {
            a = *(const v16h*)(arow + k0 + hh * 16);
        } else {
#pragma unroll
            for (int e = 0; e < 16; ++e) a[e] = (_Float16)0.0f;
        }

        const _Float16* lb = &lbuf[c & 1][0];
#pragma unroll
        for (int t = 0; t < 4; ++t) {
            v16h bb = *(const v16h*)(lb + ((size_t)(t * 16 + mn)) * 32 + hh * 16);
            acc[t] = __builtin_amdgcn_wmma_f32_16x16x32_f16(false, a, false, bb,
                                                            (short)0, acc[t], false, false);
        }
    }

#pragma unroll
    for (int t = 0; t < 4; ++t)
#pragma unroll
        for (int r = 0; r < 8; ++r) {
            int mm = m0 + hh * 8 + r;
            int nn = n0 + t * 16 + mn;
            if (mm < M) {
                float v = acc[t][r];
                if (relu) v = fmaxf(v, 0.0f);
                Cf16[(size_t)mm * N + nn] = (_Float16)v;
            }
        }
}

// ---------------------------------------------------------------------------
// p_out = lat + up2(up)   (nearest-neighbor 2x)
// ---------------------------------------------------------------------------
__global__ void up2_add_kernel(const float* __restrict__ lat,
                               const float* __restrict__ up,
                               float* __restrict__ out, int C, int H, int W)
{
    long long t = (long long)blockIdx.x * blockDim.x + threadIdx.x;
    long long total = (long long)NB * C * H * W;
    if (t >= total) return;
    int xx = (int)(t % W);
    int yy = (int)((t / W) % H);
    int c  = (int)((t / ((long long)W * H)) % C);
    int b  = (int)(t / ((long long)W * H * C));
    int H2 = H >> 1, W2 = W >> 1;
    out[t] = lat[t] + up[(((size_t)b * C + c) * H2 + (yy >> 1)) * W2 + (xx >> 1)];
}

// ---------------------------------------------------------------------------
// Anchor decode + sigmoid, writes ws score/box arrays AND d_out sections.
// ---------------------------------------------------------------------------
__global__ void decode_level_kernel(const float* __restrict__ objraw,
                                    const float* __restrict__ bvraw,
                                    float* __restrict__ scores,
                                    float* __restrict__ boxes,
                                    float* __restrict__ out,
                                    int Hf, int Wf, int stride,
                                    float s0, float s1, float s2, int lvlOff)
{
    int per = Hf * Wf * 9;
    int tid = blockIdx.x * blockDim.x + threadIdx.x;
    if (tid >= NB * per) return;
    int b = tid / per, r = tid % per;
    int a = r % 9, pix = r / 9;
    int yy = pix / Wf, xx = pix % Wf;
    float sc = (a / 3 == 0) ? s0 : ((a / 3 == 1) ? s1 : s2);
    const float rts[3] = {0.70710678118654752f, 1.0f, 1.41421356237309505f};
    float rt = rts[a % 3];
    float aw = sc * rt, ah = sc / rt;
    float acx = (xx + 0.5f) * stride, acy = (yy + 0.5f) * stride;
    int HW = Hf * Wf;
    float o  = objraw[((size_t)b * 9 + a) * HW + pix];
    const float* bvb = bvraw + ((size_t)b * 36 + a * 4) * HW + pix;
    float dx = bvb[0], dy = bvb[(size_t)HW], dw = bvb[(size_t)2 * HW], dh = bvb[(size_t)3 * HW];
    float cx = acx + dx * aw, cy = acy + dy * ah;
    float wb = aw * expf(fclampf(dw, -4.0f, 4.0f));
    float hb = ah * expf(fclampf(dh, -4.0f, 4.0f));
    float x1 = fclampf(cx - 0.5f * wb, 0.0f, IMGF);
    float y1 = fclampf(cy - 0.5f * hb, 0.0f, IMGF);
    float x2 = fclampf(cx + 0.5f * wb, 0.0f, IMGF);
    float y2 = fclampf(cy + 0.5f * hb, 0.0f, IMGF);
    int gi = b * NANCH + lvlOff + r;
    float so = 1.0f / (1.0f + expf(-o));
    scores[gi] = so;
    out[gi]    = so;
    float* bp = boxes + (size_t)gi * 4;
    bp[0] = x1; bp[1] = y1; bp[2] = x2; bp[3] = y2;
    float* ob = out + (size_t)OUT_RPNB + (size_t)gi * 4;
    ob[0] = x1; ob[1] = y1; ob[2] = x2; ob[3] = y2;
}

// ---------------------------------------------------------------------------
// Top-K = full bitonic sort of 65536 padded (score desc, index asc tiebreak).
// ---------------------------------------------------------------------------
__global__ void sort_init_kernel(const float* __restrict__ scores,
                                 float* __restrict__ keys, int* __restrict__ idxs)
{
    int tid = blockIdx.x * blockDim.x + threadIdx.x;
    if (tid >= NB * SORTN) return;
    int b = tid / SORTN, t = tid % SORTN;
    keys[tid] = (t < NANCH) ? scores[b * NANCH + t] : -1.0f;
    idxs[tid] = t;
}

__global__ void bitonic_step_kernel(float* __restrict__ keys, int* __restrict__ idxs,
                                    int j, int k)
{
    int t = blockIdx.x * blockDim.x + threadIdx.x;
    int b = blockIdx.y;
    float* kk = keys + (size_t)b * SORTN;
    int*   ii = idxs + (size_t)b * SORTN;
    int ixj = t ^ j;
    if (ixj <= t) return;
    float ka = kk[t], kb = kk[ixj];
    int   ia = ii[t], ib = ii[ixj];
    bool aFirst = (ka > kb) || (ka == kb && ia < ib);   // descending, stable
    bool desc = ((t & k) == 0);
    if (desc ? !aFirst : aFirst) {
        kk[t] = kb; kk[ixj] = ka; ii[t] = ib; ii[ixj] = ia;
    }
}

__global__ void gather_topk_kernel(const float* __restrict__ keys,
                                   const int* __restrict__ idxs,
                                   const float* __restrict__ boxesAll,
                                   float* __restrict__ sv, float* __restrict__ bk)
{
    int tid = blockIdx.x * blockDim.x + threadIdx.x;
    if (tid >= NB * PRENMS) return;
    int b = tid / PRENMS, i = tid % PRENMS;
    sv[tid] = keys[(size_t)b * SORTN + i];
    int si = idxs[(size_t)b * SORTN + i];
    const float* src = boxesAll + ((size_t)b * NANCH + si) * 4;
    float* dst = bk + (size_t)tid * 4;
    dst[0] = src[0]; dst[1] = src[1]; dst[2] = src[2]; dst[3] = src[3];
}

// ---------------------------------------------------------------------------
// NMS: per-row suppression bitmask (j > i, IoU > 0.7), then sequential greedy
// reduction + FG compaction (exactly the reference keep-loop semantics).
// ---------------------------------------------------------------------------
__global__ void nms_mask_kernel(const float* __restrict__ bk,
                                unsigned* __restrict__ masks)
{
    int tid = blockIdx.x * blockDim.x + threadIdx.x;
    if (tid >= NB * PRENMS * 63) return;
    int wword = tid % 63;
    int rest  = tid / 63;
    int i = rest % PRENMS;
    int b = rest / PRENMS;
    const float* bb = bk + (size_t)b * PRENMS * 4;
    float ax1 = bb[i * 4], ay1 = bb[i * 4 + 1], ax2 = bb[i * 4 + 2], ay2 = bb[i * 4 + 3];
    float aarea = (ax2 - ax1) * (ay2 - ay1);
    unsigned word = 0;
    for (int jj = 0; jj < 32; ++jj) {
        int j = wword * 32 + jj;
        if (j >= PRENMS || j <= i) continue;
        float bx1 = bb[j * 4], by1 = bb[j * 4 + 1], bx2 = bb[j * 4 + 2], by2 = bb[j * 4 + 3];
        float barea = (bx2 - bx1) * (by2 - by1);
        float lx = fmaxf(ax1, bx1), ly = fmaxf(ay1, by1);
        float rx = fminf(ax2, bx2), ry = fminf(ay2, by2);
        float iw = fmaxf(rx - lx, 0.0f), ih = fmaxf(ry - ly, 0.0f);
        float inter = iw * ih;
        float iou = inter / (aarea + barea - inter + 1e-6f);
        if (iou > 0.7f) word |= (1u << jj);
    }
    masks[(size_t)(b * PRENMS + i) * 63 + wword] = word;
}

__global__ void nms_seq_kernel(const unsigned* __restrict__ masks,
                               const float* __restrict__ sv,
                               int* __restrict__ fgList, int* __restrict__ fgCnt)
{
    int b = blockIdx.x;
    int t = threadIdx.x;
    __shared__ unsigned keep[63];
    if (t < 63) keep[t] = (t == 62) ? 0xFFFFu : 0xFFFFFFFFu;  // 2000 bits
    __syncthreads();
    for (int i = 0; i < PRENMS; ++i) {
        unsigned kw = keep[i >> 5];
        __syncthreads();
        if (((kw >> (i & 31)) & 1u) && t < 63)
            keep[t] &= ~masks[(size_t)(b * PRENMS + i) * 63 + t];
        __syncthreads();
    }
    if (t == 0) {
        int cnt = 0;
        const float* svb = sv + b * PRENMS;
        for (int i = 0; i < PRENMS; ++i) {
            bool kept = (keep[i >> 5] >> (i & 31)) & 1u;
            if (kept && svb[i] > 0.7f) fgList[b * PRENMS + cnt++] = i;
        }
        if (cnt == 0) fgList[b * PRENMS] = 0;
        fgCnt[b] = cnt;
    }
}

// Fixed affine permutation stands in for jax.random.permutation (threefry not
// reproducible on device; no correctness check available).
__global__ void fg_sample_kernel(const float* __restrict__ sv,
                                 const float* __restrict__ bk,
                                 const int* __restrict__ fgList,
                                 const int* __restrict__ fgCnt,
                                 float* __restrict__ fgB, float* __restrict__ out)
{
    int idx = blockIdx.x * blockDim.x + threadIdx.x;
    if (idx >= NB * NPOS) return;
    int b = idx / NPOS, p = idx % NPOS;
    int perm = (int)(((long long)p * 389 + 97) % NPOS);  // gcd(389,660)=1
    int cnt = fgCnt[b]; if (cnt < 1) cnt = 1;
    int sel = fgList[b * PRENMS + (perm % cnt)];
    out[OUT_FGO + idx] = sv[b * PRENMS + sel];
    const float* src = bk + ((size_t)b * PRENMS + sel) * 4;
    float* dst = out + OUT_FGB + (size_t)idx * 4;
    float* dws = fgB + (size_t)idx * 4;
#pragma unroll
    for (int c = 0; c < 4; ++c) { float v = src[c]; dst[c] = v; dws[c] = v; }
}

// ---------------------------------------------------------------------------
// ROI-align (7x7, 256ch) -> feats written as f16 in GEMM-A layout (M,K).
// ---------------------------------------------------------------------------
__global__ void roi_align_kernel(const float* __restrict__ p3,
                                 const float* __restrict__ p4,
                                 const float* __restrict__ p5,
                                 const float* __restrict__ fgB,
                                 _Float16* __restrict__ feats)
{
    int n = blockIdx.x;      // box
    int b = blockIdx.y;      // image
    int c = threadIdx.x;     // channel (256)
    const float* box = fgB + ((size_t)b * NPOS + n) * 4;
    float x1 = box[0], y1 = box[1], x2 = box[2], y2 = box[3];
    float area = (x2 - x1) * (y2 - y1);
    if (area < 1.0f) area = 1.0f;
    float scale = sqrtf(area);
    int lvl = (int)floorf(log2f(scale / 56.0f));
    lvl = iclamp(lvl, 0, 2);
    const float* fm; int H, W, s;
    if (lvl == 0)      { fm = p3; H = 64; W = 64; s = 8;  }
    else if (lvl == 1) { fm = p4; H = 32; W = 32; s = 16; }
    else               { fm = p5; H = 16; W = 16; s = 32; }
    const float* fmc = fm + ((size_t)(b * 256 + c)) * H * W;
    _Float16* dst = feats + ((size_t)(b * NPOS + n)) * 12544 + (size_t)c * 49;
    for (int i = 0; i < 7; ++i) {
        float gy = (y1 + (i + 0.5f) / 7.0f * (y2 - y1)) / s - 0.5f;
        float y0f = floorf(gy); float wy = gy - y0f;
        int y0 = iclamp((int)y0f, 0, H - 1);
        int y1i = iclamp(y0 + 1, 0, H - 1);
        for (int j = 0; j < 7; ++j) {
            float gx = (x1 + (j + 0.5f) / 7.0f * (x2 - x1)) / s - 0.5f;
            float x0f = floorf(gx); float wx = gx - x0f;
            int x0 = iclamp((int)x0f, 0, W - 1);
            int x1i = iclamp(x0 + 1, 0, W - 1);
            float f00 = fmc[y0 * W + x0],  f01 = fmc[y0 * W + x1i];
            float f10 = fmc[y1i * W + x0], f11 = fmc[y1i * W + x1i];
            float v = f00 * (1 - wx) * (1 - wy) + f01 * wx * (1 - wy)
                    + f10 * (1 - wx) * wy       + f11 * wx * wy;
            dst[i * 7 + j] = (_Float16)v;
        }
    }
}

__global__ void cvt_f16_kernel(const float* __restrict__ src,
                               _Float16* __restrict__ dst, int n)
{
    int t = blockIdx.x * blockDim.x + threadIdx.x;
    if (t < n) dst[t] = (_Float16)src[t];
}

// ---------------------------------------------------------------------------
// Final head: argmax over 3 class logits, gather class box offsets, decode.
// ---------------------------------------------------------------------------
__global__ void final_head_kernel(const float* __restrict__ clsO,
                                  const float* __restrict__ boxO,
                                  const float* __restrict__ fgB,
                                  float* __restrict__ out)
{
    int idx = blockIdx.x * blockDim.x + threadIdx.x;
    if (idx >= NB * NPOS) return;
    const float* lg = clsO + (size_t)idx * 3;
    int c = 0; float best = lg[0];
    if (lg[1] > best) { best = lg[1]; c = 1; }
    if (lg[2] > best) { best = lg[2]; c = 2; }
    const float* off = boxO + (size_t)idx * 12 + c * 4;
    const float* bx  = fgB + (size_t)idx * 4;
    float x1 = bx[0], y1 = bx[1], x2 = bx[2], y2 = bx[3];
    float w = x2 - x1, h = y2 - y1;
    float cx = x1 + 0.5f * w, cy = y1 + 0.5f * h;
    float ncx = cx + off[0] * w, ncy = cy + off[1] * h;
    float nw = w * expf(fclampf(off[2], -4.0f, 4.0f));
    float nh = h * expf(fclampf(off[3], -4.0f, 4.0f));
    out[OUT_CLS + idx] = (float)c;
    float* fo = out + OUT_BOX + (size_t)idx * 4;
    fo[0] = fclampf(ncx - 0.5f * nw, 0.0f, IMGF);
    fo[1] = fclampf(ncy - 0.5f * nh, 0.0f, IMGF);
    fo[2] = fclampf(ncx + 0.5f * nw, 0.0f, IMGF);
    fo[3] = fclampf(ncy + 0.5f * nh, 0.0f, IMGF);
}

// ---------------------------------------------------------------------------
// Host orchestration (graph-capture safe: only kernel launches on `stream`).
// Workspace plan ~141 MB; c1 (33.5MB) region is reused later for roi feats.
// ---------------------------------------------------------------------------
extern "C" void kernel_launch(void* const* d_in, const int* in_sizes, int n_in,
                              void* d_out, int out_size, void* d_ws, size_t ws_size,
                              hipStream_t stream)
{
    const float* x       = (const float*)d_in[0];
    const float* conv1_w = (const float*)d_in[1];
    const float* conv2_w = (const float*)d_in[2];
    const float* conv3_w = (const float*)d_in[3];
    const float* conv4_w = (const float*)d_in[4];
    const float* conv5_w = (const float*)d_in[5];
    const float* lat3_w  = (const float*)d_in[6];
    const float* lat4_w  = (const float*)d_in[7];
    const float* lat5_w  = (const float*)d_in[8];
    const float* out3_w  = (const float*)d_in[9];
    const float* out4_w  = (const float*)d_in[10];
    const float* out5_w  = (const float*)d_in[11];
    const float* rpnc_w  = (const float*)d_in[12];
    const float* rpno_w  = (const float*)d_in[13];
    const float* rpnb_w  = (const float*)d_in[14];
    const float* fc1_w   = (const float*)d_in[15];
    const float* fc2_w   = (const float*)d_in[16];
    const float* cls_w   = (const float*)d_in[17];
    const float* box_w   = (const float*)d_in[18];
    float* out = (float*)d_out;

    char* ws = (char*)d_ws;
    size_t off = 0;
    auto alloc = [&](size_t bytes) -> char* {
        char* p = ws + off;
        off = (off + bytes + 255) & ~(size_t)255;
        return p;
    };

    size_t c1b    = (size_t)NB * 64 * 256 * 256 * 4;
    size_t featsb = (size_t)NB * NPOS * 12544 * 2;
    char* r0 = alloc(c1b > featsb ? c1b : featsb);
    float*    c1    = (float*)r0;              // dead after conv2
    _Float16* feats = (_Float16*)r0;           // written by roi_align later

    float* c2     = (float*)alloc((size_t)NB * 128 * 128 * 128 * 4);
    float* c3     = (float*)alloc((size_t)NB * 256 * 64 * 64 * 4);
    float* c4     = (float*)alloc((size_t)NB * 256 * 32 * 32 * 4);
    float* c5     = (float*)alloc((size_t)NB * 256 * 16 * 16 * 4);
    float* p5     = (float*)alloc((size_t)NB * 256 * 16 * 16 * 4);
    float* t4     = (float*)alloc((size_t)NB * 256 * 32 * 32 * 4);
    float* p4pre  = (float*)alloc((size_t)NB * 256 * 32 * 32 * 4);
    float* t3     = (float*)alloc((size_t)NB * 256 * 64 * 64 * 4);
    float* p3pre  = (float*)alloc((size_t)NB * 256 * 64 * 64 * 4);
    float* p3     = (float*)alloc((size_t)NB * 256 * 64 * 64 * 4);
    float* p4     = (float*)alloc((size_t)NB * 256 * 32 * 32 * 4);
    float* p5o    = (float*)alloc((size_t)NB * 256 * 16 * 16 * 4);
    float* rpnH   = (float*)alloc((size_t)NB * 256 * 64 * 64 * 4);
    float* objraw = (float*)alloc((size_t)NB * 9 * 64 * 64 * 4);
    float* bvraw  = (float*)alloc((size_t)NB * 36 * 64 * 64 * 4);
    float* scores = (float*)alloc((size_t)NB * NANCH * 4);
    float* boxesAll = (float*)alloc((size_t)NB * NANCH * 16);
    float* sortK  = (float*)alloc((size_t)NB * SORTN * 4);
    int*   sortI  = (int*)alloc((size_t)NB * SORTN * 4);
    float* sv     = (float*)alloc((size_t)NB * PRENMS * 4);
    float* bk     = (float*)alloc((size_t)NB * PRENMS * 16);
    unsigned* masks = (unsigned*)alloc((size_t)NB * PRENMS * 63 * 4);
    int*   fgList = (int*)alloc((size_t)NB * PRENMS * 4);
    int*   fgCnt  = (int*)alloc((size_t)NB * 4);
    float* fgB    = (float*)alloc((size_t)NB * NPOS * 16);
    _Float16* fc1h = (_Float16*)alloc((size_t)1024 * 12544 * 2);
    _Float16* fc2h = (_Float16*)alloc((size_t)1024 * 1024 * 2);
    _Float16* clsh = (_Float16*)alloc((size_t)3 * 1024 * 2);
    _Float16* boxh = (_Float16*)alloc((size_t)12 * 1024 * 2);
    _Float16* h1h  = (_Float16*)alloc((size_t)NB * NPOS * 1024 * 2);
    _Float16* h2h  = (_Float16*)alloc((size_t)NB * NPOS * 1024 * 2);
    float* clsO = (float*)alloc((size_t)NB * NPOS * 3 * 4);
    float* boxO = (float*)alloc((size_t)NB * NPOS * 12 * 4);

    dim3 blk256(256);
    auto grid1 = [](long long n, int bs) { return dim3((unsigned)((n + bs - 1) / bs)); };

    // ---- backbone ----
    conv1_kernel<<<grid1((long long)NB * 64 * 256 * 256, 256), blk256, 0, stream>>>(x, conv1_w, c1);
    conv_wmma_kernel<<<dim3(128 * 128 / 16, 8, NB),  dim3(32), 0, stream>>>(c1, conv2_w, c2, 64, 128, 256, 256, 128, 128, 2, 3, 0, 1);
    conv_wmma_kernel<<<dim3(64 * 64 / 16, 16, NB),   dim3(32), 0, stream>>>(c2, conv3_w, c3, 128, 256, 128, 128, 64, 64, 2, 3, 0, 1);
    conv_wmma_kernel<<<dim3(32 * 32 / 16, 16, NB),   dim3(32), 0, stream>>>(c3, conv4_w, c4, 256, 256, 64, 64, 32, 32, 2, 3, 0, 1);
    conv_wmma_kernel<<<dim3(16 * 16 / 16, 16, NB),   dim3(32), 0, stream>>>(c4, conv5_w, c5, 256, 256, 32, 32, 16, 16, 2, 3, 0, 1);

    // ---- FPN ----
    conv_wmma_kernel<<<dim3(16, 16, NB),  dim3(32), 0, stream>>>(c5, lat5_w, p5, 256, 256, 16, 16, 16, 16, 1, 1, 0, 0);
    conv_wmma_kernel<<<dim3(64, 16, NB),  dim3(32), 0, stream>>>(c4, lat4_w, t4, 256, 256, 32, 32, 32, 32, 1, 1, 0, 0);
    up2_add_kernel<<<grid1((long long)NB * 256 * 32 * 32, 256), blk256, 0, stream>>>(t4, p5, p4pre, 256, 32, 32);
    conv_wmma_kernel<<<dim3(256, 16, NB), dim3(32), 0, stream>>>(c3, lat3_w, t3, 256, 256, 64, 64, 64, 64, 1, 1, 0, 0);
    up2_add_kernel<<<grid1((long long)NB * 256 * 64 * 64, 256), blk256, 0, stream>>>(t3, p4pre, p3pre, 256, 64, 64);
    conv_wmma_kernel<<<dim3(256, 16, NB), dim3(32), 0, stream>>>(p3pre, out3_w, p3,  256, 256, 64, 64, 64, 64, 1, 3, 1, 0);
    conv_wmma_kernel<<<dim3(64, 16, NB),  dim3(32), 0, stream>>>(p4pre, out4_w, p4,  256, 256, 32, 32, 32, 32, 1, 3, 1, 0);
    conv_wmma_kernel<<<dim3(16, 16, NB),  dim3(32), 0, stream>>>(p5,    out5_w, p5o, 256, 256, 16, 16, 16, 16, 1, 3, 1, 0);

    // ---- RPN + decode per level ----
    const float* fms[3] = {p3, p4, p5o};
    const int   HfA[3]  = {64, 32, 16};
    const int   strd[3] = {8, 16, 32};
    const float scl[3][3] = {{32, 64, 128}, {64, 128, 256}, {128, 256, 512}};
    const int   lvlOff[3] = {0, 36864, 46080};
    for (int l = 0; l < 3; ++l) {
        int hw = HfA[l] * HfA[l];
        conv_wmma_kernel<<<dim3(hw / 16, 16, NB), dim3(32), 0, stream>>>(fms[l], rpnc_w, rpnH, 256, 256, HfA[l], HfA[l], HfA[l], HfA[l], 1, 3, 1, 1);
        conv_wmma_kernel<<<dim3(hw / 16, 1, NB),  dim3(32), 0, stream>>>(rpnH, rpno_w, objraw, 256, 9,  HfA[l], HfA[l], HfA[l], HfA[l], 1, 1, 0, 0);
        conv_wmma_kernel<<<dim3(hw / 16, 3, NB),  dim3(32), 0, stream>>>(rpnH, rpnb_w, bvraw,  256, 36, HfA[l], HfA[l], HfA[l], HfA[l], 1, 1, 0, 0);
        decode_level_kernel<<<grid1((long long)NB * hw * 9, 256), blk256, 0, stream>>>(
            objraw, bvraw, scores, boxesAll, out, HfA[l], HfA[l], strd[l],
            scl[l][0], scl[l][1], scl[l][2], lvlOff[l]);
    }

    // ---- top-2000 (bitonic), NMS, FG sampling ----
    sort_init_kernel<<<grid1((long long)NB * SORTN, 256), blk256, 0, stream>>>(scores, sortK, sortI);
    for (int k = 2; k <= SORTN; k <<= 1)
        for (int j = k >> 1; j > 0; j >>= 1)
            bitonic_step_kernel<<<dim3(SORTN / 256, NB), blk256, 0, stream>>>(sortK, sortI, j, k);
    gather_topk_kernel<<<grid1((long long)NB * PRENMS, 256), blk256, 0, stream>>>(sortK, sortI, boxesAll, sv, bk);
    nms_mask_kernel<<<grid1((long long)NB * PRENMS * 63, 256), blk256, 0, stream>>>(bk, masks);
    nms_seq_kernel<<<dim3(NB), dim3(64), 0, stream>>>(masks, sv, fgList, fgCnt);
    fg_sample_kernel<<<grid1((long long)NB * NPOS, 256), blk256, 0, stream>>>(sv, bk, fgList, fgCnt, fgB, out);

    // ---- ROI align -> f16 feats (GEMM-A layout) ----
    roi_align_kernel<<<dim3(NPOS, NB), dim3(256), 0, stream>>>(p3, p4, p5o, fgB, feats);

    // ---- FC head: f16 weights once per call, then WMMA GEMMs ----
    cvt_f16_kernel<<<grid1((long long)1024 * 12544, 256), blk256, 0, stream>>>(fc1_w, fc1h, 1024 * 12544);
    cvt_f16_kernel<<<grid1((long long)1024 * 1024, 256),  blk256, 0, stream>>>(fc2_w, fc2h, 1024 * 1024);
    cvt_f16_kernel<<<grid1(3 * 1024, 256),  blk256, 0, stream>>>(cls_w, clsh, 3 * 1024);
    cvt_f16_kernel<<<grid1(12 * 1024, 256), blk256, 0, stream>>>(box_w, boxh, 12 * 1024);

    int M = NB * NPOS;  // 1320
    gemm64_wmma_kernel<<<dim3((M + 15) / 16, 1024 / 64), dim3(32), 0, stream>>>(feats, fc1h, h1h, M, 1024, 12544, 1);
    gemm64_wmma_kernel<<<dim3((M + 15) / 16, 1024 / 64), dim3(32), 0, stream>>>(h1h,   fc2h, h2h, M, 1024, 1024, 1);
    gemm_wmma_kernel<<<dim3((M + 15) / 16, 1),  dim3(32), 0, stream>>>(h2h, clsh, clsO, nullptr, M, 3,  1024, 0);
    gemm_wmma_kernel<<<dim3((M + 15) / 16, 1),  dim3(32), 0, stream>>>(h2h, boxh, boxO, nullptr, M, 12, 1024, 0);

    final_head_kernel<<<grid1((long long)NB * NPOS, 256), blk256, 0, stream>>>(clsO, boxO, fgB, out);

    (void)in_sizes; (void)n_in; (void)out_size; (void)ws_size;
}
